// _LoraExpert_76003741270246
// MI455X (gfx1250) — matrix operand order repaired and checked by reading, not compiled, not run on hardware
//
#include <hip/hip_runtime.h>

// ---------------------------------------------------------------------------
// MoE expert MLP:  y = gelu(x @ W1^T + b1) @ W2^T + b2, 64 experts x 64 tokens.
// Weight streaming (2.15 GB fp32, read exactly once) dominates -> memory bound
// at ~93 us on 23.3 TB/s.  Compute path: fp32->bf16 on the fly, bf16 WMMA with
// fp32 accumulation (v_wmma_f32_16x16x32_bf16).  The 32 MB bf16 intermediate
// stays in the 192 MB L2; the weight stream uses non-temporal loads so the
// one-shot 2.15 GB sweep does not evict it.  Kernel 2 stages the bf16 h via
// the CDNA5 async-to-LDS engine (ASYNCcnt) with double buffering.
// ---------------------------------------------------------------------------

typedef __attribute__((ext_vector_type(16))) __bf16       v16bf;
typedef __attribute__((ext_vector_type(8)))  float        v8f;
typedef __attribute__((ext_vector_type(4)))  float        f32x4;
typedef __attribute__((ext_vector_type(4)))  unsigned int u32x4;
typedef __attribute__((ext_vector_type(4)))  int          i32x4;

#define NE   64          // experts
#define ND   1024        // d_model
#define NH   4096        // d_hidden
#define TPE  64          // tokens per expert (uniform counts)
#define LDS1_STRIDE (ND + 8)   // k1 LDS row stride (bf16 elems), bank de-conflict
#define BK2  256               // k2 K-chunk staged in LDS
#define LDS2_STRIDE (BK2 + 8)  // k2 LDS row stride (bf16 elems)

#if defined(__HIP_DEVICE_COMPILE__) && \
    __has_builtin(__builtin_amdgcn_global_load_async_to_lds_b128) && \
    __has_builtin(__builtin_amdgcn_s_wait_asynccnt)
#define USE_ASYNC_COPY 1
typedef __attribute__((address_space(1))) i32x4 gi32x4;   // global int4
typedef __attribute__((address_space(3))) i32x4 li32x4;   // LDS int4
#else
#define USE_ASYNC_COPY 0
#endif

union FragBF {           // one 16x32 (or 32x16) bf16 WMMA operand
  u32x4 q[2];            // two 16-byte halves (8 bf16 each)
  v16bf v;               // WMMA operand view
};

__device__ __forceinline__ v8f wmma_bf16(v16bf a, v16bf b, v8f c) {
  // D = A(16x32) * B(32x16) + C, fp32 accumulate
  return __builtin_amdgcn_wmma_f32_16x16x32_bf16(
      /*neg_a=*/false, a, /*neg_b=*/false, b,
      /*c_mod=*/(short)0, c, /*reuse_a=*/false, /*reuse_b=*/false);
}

// Non-temporal fp32x4 weight load (TH=NT: one-shot stream, keep out of L2)
__device__ __forceinline__ f32x4 ld_stream(const float* p) {
  return __builtin_nontemporal_load((const f32x4*)p);
}

// ---------------------------------------------------------------------------
// Kernel 1: h = gelu(X @ W1^T + b1), stored bf16 into scratch.
// grid = (NH/256, NE), block = 256 (8 waves).  Wave w owns 32 N-columns.
// X_e staged fp32->bf16 in LDS; W1 streamed (NT) from HBM exactly once.
// ---------------------------------------------------------------------------
extern "C" __global__ __launch_bounds__(256)
void moe_htoh4(const float* __restrict__ inp, const float* __restrict__ w1,
               const float* __restrict__ b1, __bf16* __restrict__ hbuf)
{
  extern __shared__ __bf16 Xs[];            // [TPE][LDS1_STRIDE] = 132 KB
  const int e   = blockIdx.y;
  const int n0  = blockIdx.x * 256;
  const int tid = threadIdx.x;

  // Stage this expert's activations (64 x 1024 fp32 -> bf16) into LDS.
  const float* xe = inp + (size_t)e * TPE * ND;
  for (int i = tid; i < TPE * ND / 4; i += 256) {
    const int m = i >> 8;                   // (i*4) / 1024
    const int k = (i << 2) & (ND - 1);
    f32x4 f = *(const f32x4*)(xe + m * ND + k);
    __bf16* d = &Xs[m * LDS1_STRIDE + k];
    d[0] = (__bf16)f[0]; d[1] = (__bf16)f[1]; d[2] = (__bf16)f[2]; d[3] = (__bf16)f[3];
  }
  __syncthreads();

  const int wave = tid >> 5, lane = tid & 31;
  const int lhi = lane >> 4, llo = lane & 15;
  const int nb  = n0 + wave * 32;           // this wave's first output column

  v8f acc[2][4] = {};                       // [n-tile][m-tile] fp32 accumulators

  const float* w1e = w1 + (size_t)e * NH * ND;
  for (int kk = 0; kk < ND; kk += 32) {
    // A fragments: lane holds K in {ka..ka+15} (contiguous, matches B)
    FragBF a[4];
    const int ka = kk + lhi * 16;
    #pragma unroll
    for (int mt = 0; mt < 4; ++mt) {
      const __bf16* s = &Xs[(mt * 16 + llo) * LDS1_STRIDE + ka];
      a[mt].q[0] = *(const u32x4*)(s);       // K = ka .. ka+7
      a[mt].q[1] = *(const u32x4*)(s + 8);   // K = ka+8 .. ka+15
    }
    // B fragments: lane = column n (llo), K = kk + lhi*16 .. +15; fp32 -> bf16
    FragBF bb[2];
    #pragma unroll
    for (int nt = 0; nt < 2; ++nt) {
      const float* wr = w1e + (size_t)(nb + nt * 16 + llo) * ND + kk + lhi * 16;
      #pragma unroll
      for (int c = 0; c < 4; ++c) {
        f32x4 f = ld_stream(wr + c * 4);
        bb[nt].v[c * 4 + 0] = (__bf16)f[0];  bb[nt].v[c * 4 + 1] = (__bf16)f[1];
        bb[nt].v[c * 4 + 2] = (__bf16)f[2];  bb[nt].v[c * 4 + 3] = (__bf16)f[3];
      }
    }
    #pragma unroll
    for (int nt = 0; nt < 2; ++nt)
      #pragma unroll
      for (int mt = 0; mt < 4; ++mt)
        acc[nt][mt] = wmma_bf16(a[mt].v, bb[nt].v, acc[nt][mt]);
  }

  // Epilogue: bias + exact gelu (erf), store bf16.  C/D layout: VGPR r ->
  // M = mt*16 + lhi*8 + r, N = tile + llo.
  #pragma unroll
  for (int nt = 0; nt < 2; ++nt) {
    const int n = nb + nt * 16 + llo;
    const float bias = b1[e * NH + n];
    #pragma unroll
    for (int mt = 0; mt < 4; ++mt) {
      #pragma unroll
      for (int r = 0; r < 8; ++r) {
        const int m = mt * 16 + lhi * 8 + r;
        float x = acc[nt][mt][r] + bias;
        float g = 0.5f * x * (1.0f + erff(x * 0.70710678118654752f));
        hbuf[((size_t)(e * TPE + m)) * NH + n] = (__bf16)g;
      }
    }
  }
}

// ---------------------------------------------------------------------------
// Kernel 2: y = h @ W2^T + b2 (fp32 out).
// grid = (ND/128, NE), block = 256 (8 waves).  Wave owns 16 N-columns; all 8
// waves share the A operand (64 x BK2 bf16 chunk), staged into a double-
// buffered LDS tile by GLOBAL_LOAD_ASYNC_TO_LDS_B128 (ASYNCcnt) and overlapped
// with WMMA on the previous chunk.  W2 streamed (NT) from HBM exactly once.
// ---------------------------------------------------------------------------

// Issue the async (or fallback) copy of one 64 x BK2 bf16 chunk into LDS.
// 64 rows * 512 B = 2048 16-byte packets; 256 threads x 8 packets each.
__device__ __forceinline__ void stage_chunk(const __bf16* he, int kk,
                                            __bf16* buf, int tid)
{
  #pragma unroll
  for (int j = 0; j < 8; ++j) {
    const int i   = j * 256 + tid;      // 16B packet index
    const int row = i >> 5;             // 32 packets per row
    const int c16 = i & 31;
    const __bf16* g = he + (size_t)row * NH + kk + c16 * 8;
    __bf16*       l = buf + row * LDS2_STRIDE + c16 * 8;
#if USE_ASYNC_COPY
    void* gv = (void*)g;                // drop const, stay generic
    void* lv = (void*)l;
    __builtin_amdgcn_global_load_async_to_lds_b128(
        (gi32x4*)gv, (li32x4*)lv, 0, 0);
#else
    *(u32x4*)l = *(const u32x4*)g;
#endif
  }
}

#if USE_ASYNC_COPY
#define WAIT_ASYNC(n) __builtin_amdgcn_s_wait_asynccnt(n)
#else
#define WAIT_ASYNC(n) ((void)0)
#endif

extern "C" __global__ __launch_bounds__(256)
void moe_h4toh(const __bf16* __restrict__ hbuf, const float* __restrict__ w2,
               const float* __restrict__ b2, float* __restrict__ out)
{
  extern __shared__ __bf16 Hs[];          // 2 x [TPE][LDS2_STRIDE] = 66 KB
  const int e    = blockIdx.y;
  const int tid  = threadIdx.x;
  const int wave = tid >> 5, lane = tid & 31;
  const int lhi  = lane >> 4, llo = lane & 15;
  const int n    = blockIdx.x * 128 + wave * 16 + llo;   // per-lane column

  v8f acc[4] = {};

  const __bf16* he  = hbuf + (size_t)e * TPE * NH;
  const float*  wr0 = w2 + (size_t)e * ND * NH + (size_t)n * NH;

  const int NCH = NH / BK2;               // 16 chunks
  stage_chunk(he, 0, Hs, tid);            // prime the pipeline

  for (int c = 0; c < NCH; ++c) {
    __bf16* cur = Hs + (c & 1) * (TPE * LDS2_STRIDE);
    if (c + 1 < NCH) {                    // prefetch next chunk into other buf
      stage_chunk(he, (c + 1) * BK2, Hs + ((c + 1) & 1) * (TPE * LDS2_STRIDE), tid);
      WAIT_ASYNC(8);                      // chunk c retired (in-order), c+1 in flight
    } else {
      WAIT_ASYNC(0);
    }
    __syncthreads();                      // all waves' packets visible in LDS

    const int kk0 = c * BK2;
    for (int kk2 = 0; kk2 < BK2; kk2 += 32) {
      FragBF a[4];
      const int ka = kk2 + lhi * 16;
      #pragma unroll
      for (int mt = 0; mt < 4; ++mt) {
        const __bf16* s = cur + (mt * 16 + llo) * LDS2_STRIDE + ka;
        a[mt].q[0] = *(const u32x4*)(s);
        a[mt].q[1] = *(const u32x4*)(s + 8);
      }
      FragBF bbf;
      const float* wr = wr0 + kk0 + kk2 + lhi * 16;
      #pragma unroll
      for (int cc = 0; cc < 4; ++cc) {
        f32x4 f = ld_stream(wr + cc * 4);
        bbf.v[cc * 4 + 0] = (__bf16)f[0];  bbf.v[cc * 4 + 1] = (__bf16)f[1];
        bbf.v[cc * 4 + 2] = (__bf16)f[2];  bbf.v[cc * 4 + 3] = (__bf16)f[3];
      }
      #pragma unroll
      for (int mt = 0; mt < 4; ++mt)
        acc[mt] = wmma_bf16(a[mt].v, bbf.v, acc[mt]);
    }
    __syncthreads();                      // waves done reading before overwrite
  }

  const float bias = b2[e * ND + n];
  #pragma unroll
  for (int mt = 0; mt < 4; ++mt) {
    #pragma unroll
    for (int r = 0; r < 8; ++r) {
      const int m = mt * 16 + lhi * 8 + r;
      out[((size_t)(e * TPE + m)) * ND + n] = acc[mt][r] + bias;
    }
  }
}

// ---------------------------------------------------------------------------
extern "C" void kernel_launch(void* const* d_in, const int* in_sizes, int n_in,
                              void* d_out, int out_size, void* d_ws, size_t ws_size,
                              hipStream_t stream)
{
  (void)in_sizes; (void)n_in; (void)out_size; (void)ws_size;
  const float* inp = (const float*)d_in[0];
  const float* w1  = (const float*)d_in[1];
  const float* b1  = (const float*)d_in[2];
  const float* w2  = (const float*)d_in[3];
  const float* b2  = (const float*)d_in[4];
  // d_in[5] = fwd_expert_count: uniform T/E per the reference; baked into grid.

  __bf16* hbuf = (__bf16*)d_ws;   // 4096*4096*2 = 32 MB scratch (L2-resident)

  const size_t lds1 = (size_t)TPE * LDS1_STRIDE * sizeof(__bf16);      // 132 KB
  const size_t lds2 = (size_t)2 * TPE * LDS2_STRIDE * sizeof(__bf16);  //  66 KB
  moe_htoh4<<<dim3(NH / 256, NE), 256, lds1, stream>>>(inp, w1, b1, hbuf);
  moe_h4toh<<<dim3(ND / 128, NE), 256, lds2, stream>>>(hbuf, w2, b2, (float*)d_out);
}
